// DispersionLoss_41772851921541
// MI455X (gfx1250) — matrix operand order repaired and checked by reading, not compiled, not run on hardware
//
#include <hip/hip_runtime.h>

typedef __attribute__((ext_vector_type(16))) _Float16 v16h;
typedef __attribute__((ext_vector_type(4)))  _Float16 v4h;
typedef __attribute__((ext_vector_type(8)))  float    v8f;
typedef __attribute__((ext_vector_type(4)))  int      v4i;

#define B_   8
#define L_   2048
#define F_   256
#define TAU1 50.0f

// ---- async double-buffered kernel tiling ----
#define BM   128
#define BN   128
#define BK   32
#define BKP  40   // padded row stride in halves (80B rows -> conflict-free b128 reads)

// ---- fallback kernel tiling (round-1 style, f32 on the fly) ----
#define FBK  64
#define FBKP 72

#if defined(__AMDGCN__) && __has_builtin(__builtin_amdgcn_global_load_async_to_lds_b128)
#define HAVE_ASYNC 1
#else
#define HAVE_ASYNC 0
#endif

#define GLOBAL_AS __attribute__((address_space(1)))
#define LDS_AS    __attribute__((address_space(3)))

typedef GLOBAL_AS v4i* gv4i_ptr;   // pointer to v4i in global (target AS 1)
typedef LDS_AS    v4i* lv4i_ptr;   // pointer to v4i in LDS    (target AS 3)

__device__ __forceinline__ void async_copy_b128(const _Float16* g, _Float16* l) {
#if HAVE_ASYNC
    __builtin_amdgcn_global_load_async_to_lds_b128((gv4i_ptr)g, (lv4i_ptr)l, 0, 0);
#else
    *(uint4*)l = *(const uint4*)g;
#endif
}

template <int N>
__device__ __forceinline__ void async_wait() {
#if HAVE_ASYNC
#if __has_builtin(__builtin_amdgcn_s_wait_asynccnt)
    __builtin_amdgcn_s_wait_asynccnt(N);
#else
    asm volatile("s_wait_asynccnt %0" :: "i"(N) : "memory");
#endif
#endif
}

__global__ void zero_accum(float* accum) {
    if (threadIdx.x == 0) accum[0] = 0.0f;
}

// f32 -> f16 copy of z (once, instead of per-tile VALU conversion)
__global__ __launch_bounds__(256) void convert_f16(const float* __restrict__ z,
                                                   _Float16* __restrict__ zh) {
    size_t i = ((size_t)blockIdx.x * 256 + threadIdx.x) * 4;
    float4 v = *(const float4*)(z + i);
    v4h h = { (_Float16)v.x, (_Float16)v.y, (_Float16)v.z, (_Float16)v.w };
    *(v4h*)(zh + i) = h;
}

// One wave (32 lanes) per row: f32 squared norm.
__global__ __launch_bounds__(256) void row_norms(const float* __restrict__ z,
                                                 float* __restrict__ norms) {
    int w    = threadIdx.x >> 5;
    int lane = threadIdx.x & 31;
    int row  = blockIdx.x * 8 + w;
    const float4* p = (const float4*)(z + (size_t)row * F_ + lane * 8);
    float4 a = p[0];
    float4 b = p[1];
    float s = a.x*a.x + a.y*a.y + a.z*a.z + a.w*a.w
            + b.x*b.x + b.y*b.y + b.z*b.z + b.w*b.w;
    #pragma unroll
    for (int m = 16; m >= 1; m >>= 1) s += __shfl_xor(s, m, 32);
    if (lane == 0) norms[row] = s;
}

// ======================= async double-buffered main kernel =======================
__global__ __launch_bounds__(256) void gram_exp_async(const _Float16* __restrict__ zh,
                                                      const float* __restrict__ norms,
                                                      float* __restrict__ accum) {
    __shared__ _Float16 As[2][BM * BKP];
    __shared__ _Float16 Bs[2][BN * BKP];
    __shared__ float    nR[BM];
    __shared__ float    nC[BN];
    __shared__ float    red[256];

    const int t  = threadIdx.x;
    const int b  = blockIdx.z;
    const int br = blockIdx.y;
    const int bc = blockIdx.x;

    const size_t zb      = (size_t)b * L_ * F_;
    const int    rowBase = br * BM;
    const int    colBase = bc * BN;

    if (t < BM) nR[t]      = norms[b * L_ + rowBase + t];
    else        nC[t - BM] = norms[b * L_ + colBase + (t - BM)];

    const int w    = t >> 5;
    const int lane = t & 31;
    const int wm   = w & 3;        // 4 row groups of 32
    const int wn   = w >> 2;       // 2 col groups of 64
    const int hi   = lane >> 4;
    const int lo16 = lane & 15;

    // panel fill: 128 rows x 32 halves = 512 b128 chunks per panel, 2/thread/panel
    const int pr0 = t >> 2;                 // row for it=0  (0..63)
    const int pq  = (t & 3) * 8;            // half-offset within row

    auto issue = [&](int k0, int buf) {
        #pragma unroll
        for (int it = 0; it < 2; ++it) {
            int r = pr0 + it * 64;
            const _Float16* ga = zh + zb + (size_t)(rowBase + r) * F_ + k0 + pq;
            async_copy_b128(ga, &As[buf][r * BKP + pq]);
            const _Float16* gb = zh + zb + (size_t)(colBase + r) * F_ + k0 + pq;
            async_copy_b128(gb, &Bs[buf][r * BKP + pq]);
        }
    };

    v8f acc[2][4] = {};

    constexpr int NC = F_ / BK;   // 8 chunks
    issue(0, 0);

    for (int c = 0; c < NC; ++c) {
        const int cur = c & 1;
        if (c + 1 < NC) issue((c + 1) * BK, cur ^ 1);
        if (c + 1 < NC) async_wait<4>();    // chunk c complete; c+1 still in flight
        else            async_wait<0>();
        __syncthreads();

        v16h afr[2], bfr[4];
        // A fragment: lane = row M; lanes 0-15 K=[0,8)+[16,24), lanes 16-31 K=[8,16)+[24,32)
        #pragma unroll
        for (int tm = 0; tm < 2; ++tm) {
            int m = wm * 32 + tm * 16 + lo16;
            const _Float16* p = &As[cur][m * BKP + hi * 8];
            union { v16h v; uint4 u[2]; } ua;
            ua.u[0] = *(const uint4*)(p);
            ua.u[1] = *(const uint4*)(p + 16);
            afr[tm] = ua.v;
        }
        // B fragment: lane = column N; lanes 0-15 K=[0,8)+[16,24), lanes 16-31 K=[8,16)+[24,32)
        #pragma unroll
        for (int tn = 0; tn < 4; ++tn) {
            int n = wn * 64 + tn * 16 + lo16;
            const _Float16* p = &Bs[cur][n * BKP + hi * 8];
            union { v16h v; uint4 u[2]; } ub;
            ub.u[0] = *(const uint4*)(p);
            ub.u[1] = *(const uint4*)(p + 16);
            bfr[tn] = ub.v;
        }
        #pragma unroll
        for (int tm = 0; tm < 2; ++tm)
            #pragma unroll
            for (int tn = 0; tn < 4; ++tn)
                acc[tm][tn] = __builtin_amdgcn_wmma_f32_16x16x32_f16(
                    false, afr[tm], false, bfr[tn],
                    (short)0, acc[tm][tn], false, false);
        __syncthreads();
    }

    // ---- fused epilogue: sqd -> exp -> masked partial sum ----
    const float invLT = 1.0f / ((float)L_ * TAU1);
    float partial = 0.0f;
    #pragma unroll
    for (int tm = 0; tm < 2; ++tm) {
        #pragma unroll
        for (int tn = 0; tn < 4; ++tn) {
            const int rb   = wm * 32 + tm * 16 + hi * 8;
            const int cb   = wn * 64 + tn * 16 + lo16;
            const int gcol = colBase + cb;
            const float nc = nC[cb];
            #pragma unroll
            for (int i = 0; i < 8; ++i) {
                const int r = rb + i;
                float sqd = nR[r] + nc - 2.0f * acc[tm][tn][i];
                sqd = fmaxf(sqd, 0.0f);
                float e = __expf(-sqd * invLT);
                partial += ((rowBase + r) != gcol) ? e : 0.0f;
            }
        }
    }

    red[t] = partial;
    __syncthreads();
    #pragma unroll
    for (int s = 128; s > 0; s >>= 1) {
        if (t < s) red[t] += red[t + s];
        __syncthreads();
    }
    if (t == 0) atomicAdd(accum, red[0]);
}

// ======================= fallback kernel (small ws): f32 -> f16 on the fly =======================
__global__ __launch_bounds__(256) void gram_exp(const float* __restrict__ z,
                                                const float* __restrict__ norms,
                                                float* __restrict__ accum) {
    __shared__ _Float16 As[BM * FBKP];
    __shared__ _Float16 Bs[BN * FBKP];
    __shared__ float    nR[BM];
    __shared__ float    nC[BN];
    __shared__ float    red[256];

    const int t  = threadIdx.x;
    const int b  = blockIdx.z;
    const int br = blockIdx.y;
    const int bc = blockIdx.x;

    const size_t zb      = (size_t)b * L_ * F_;
    const int    rowBase = br * BM;
    const int    colBase = bc * BN;

    if (t < BM) nR[t]      = norms[b * L_ + rowBase + t];
    else        nC[t - BM] = norms[b * L_ + colBase + (t - BM)];

    const int w    = t >> 5;
    const int lane = t & 31;
    const int wm   = w & 3;
    const int wn   = w >> 2;
    const int hi   = lane >> 4;
    const int lo16 = lane & 15;

    v8f acc[2][4] = {};

    for (int k0 = 0; k0 < F_; k0 += FBK) {
        #pragma unroll
        for (int it = 0; it < 8; ++it) {
            int idx = it * 256 + t;
            int r   = idx >> 4;
            int q   = idx & 15;
            float4 va = ((const float4*)(z + zb + (size_t)(rowBase + r) * F_ + k0))[q];
            v4h ha = { (_Float16)va.x, (_Float16)va.y, (_Float16)va.z, (_Float16)va.w };
            *(v4h*)(As + r * FBKP + q * 4) = ha;
            float4 vb = ((const float4*)(z + zb + (size_t)(colBase + r) * F_ + k0))[q];
            v4h hb = { (_Float16)vb.x, (_Float16)vb.y, (_Float16)vb.z, (_Float16)vb.w };
            *(v4h*)(Bs + r * FBKP + q * 4) = hb;
        }
        __syncthreads();

        #pragma unroll
        for (int ki = 0; ki < 2; ++ki) {
            const int kb = ki * 32;
            v16h afr[2], bfr[4];
            #pragma unroll
            for (int tm = 0; tm < 2; ++tm) {
                int m = wm * 32 + tm * 16 + lo16;
                const _Float16* p = As + m * FBKP + kb + hi * 8;
                union { v16h v; uint4 u[2]; } ua;
                ua.u[0] = *(const uint4*)(p);
                ua.u[1] = *(const uint4*)(p + 16);
                afr[tm] = ua.v;
            }
            #pragma unroll
            for (int tn = 0; tn < 4; ++tn) {
                int n = wn * 64 + tn * 16 + lo16;
                const _Float16* p = Bs + n * FBKP + kb + hi * 8;
                union { v16h v; uint4 u[2]; } ub;
                ub.u[0] = *(const uint4*)(p);
                ub.u[1] = *(const uint4*)(p + 16);
                bfr[tn] = ub.v;
            }
            #pragma unroll
            for (int tm = 0; tm < 2; ++tm)
                #pragma unroll
                for (int tn = 0; tn < 4; ++tn)
                    acc[tm][tn] = __builtin_amdgcn_wmma_f32_16x16x32_f16(
                        false, afr[tm], false, bfr[tn],
                        (short)0, acc[tm][tn], false, false);
        }
        __syncthreads();
    }

    const float invLT = 1.0f / ((float)L_ * TAU1);
    float partial = 0.0f;
    #pragma unroll
    for (int tm = 0; tm < 2; ++tm) {
        #pragma unroll
        for (int tn = 0; tn < 4; ++tn) {
            const int rb   = wm * 32 + tm * 16 + hi * 8;
            const int cb   = wn * 64 + tn * 16 + lo16;
            const int gcol = colBase + cb;
            const float nc = nC[cb];
            #pragma unroll
            for (int i = 0; i < 8; ++i) {
                const int r = rb + i;
                float sqd = nR[r] + nc - 2.0f * acc[tm][tn][i];
                sqd = fmaxf(sqd, 0.0f);
                float e = __expf(-sqd * invLT);
                partial += ((rowBase + r) != gcol) ? e : 0.0f;
            }
        }
    }

    red[t] = partial;
    __syncthreads();
    #pragma unroll
    for (int s = 128; s > 0; s >>= 1) {
        if (t < s) red[t] += red[t + s];
        __syncthreads();
    }
    if (t == 0) atomicAdd(accum, red[0]);
}

__global__ void finalize(const float* __restrict__ accum, float* __restrict__ out) {
    if (threadIdx.x == 0) {
        const float count = (float)B_ * (float)L_ * (float)(L_ - 1);
        out[0] = logf(accum[0] / count);
    }
}

extern "C" void kernel_launch(void* const* d_in, const int* in_sizes, int n_in,
                              void* d_out, int out_size, void* d_ws, size_t ws_size,
                              hipStream_t stream) {
    const float* z     = (const float*)d_in[0];
    float*       accum = (float*)d_ws;                       // 1 float  @ 0
    float*       norms = (float*)((char*)d_ws + 16);         // B_*L_ floats @ 16
    _Float16*    zh    = (_Float16*)((char*)d_ws + 16 + sizeof(float) * B_ * L_);
    float*       out   = (float*)d_out;

    const size_t need_f16 = 16 + sizeof(float) * (size_t)B_ * L_
                          + sizeof(_Float16) * (size_t)B_ * L_ * F_;

    zero_accum<<<1, 64, 0, stream>>>(accum);
    row_norms<<<(B_ * L_) / 8, 256, 0, stream>>>(z, norms);

    dim3 grid(L_ / BN, L_ / BM, B_);
    if (ws_size >= need_f16) {
        convert_f16<<<(B_ * L_ * F_) / (256 * 4), 256, 0, stream>>>(z, zh);
        gram_exp_async<<<grid, 256, 0, stream>>>(zh, norms, accum);
    } else {
        gram_exp<<<grid, 256, 0, stream>>>(z, norms, accum);
    }
    finalize<<<1, 32, 0, stream>>>(accum, out);
}
